// PHMExpert_26680336843014
// MI455X (gfx1250) — matrix operand order repaired and checked by reading, not compiled, not run on hardware
//
#include <hip/hip_runtime.h>
#include <hip/hip_bf16.h>

typedef __attribute__((ext_vector_type(16))) __bf16 v16bf;
typedef __attribute__((ext_vector_type(8)))  float  v8f;

#define T_TOK 8192
#define DIM   2048
#define ODIM  2048
#define NEXP  8

// ---- workspace layout (bytes) ----
#define WT_OFF     ((size_t)0)
#define WT_BYTES   ((size_t)NEXP * DIM * ODIM * 2)      // 64 MiB bf16 WfullT [e][o][d]
#define XB_OFF     (WT_OFF + WT_BYTES)
#define XB_BYTES   ((size_t)T_TOK * DIM * 2)            // 32 MiB bf16 x
#define PERM_OFF   (XB_OFF + XB_BYTES)
#define PERM_BYTES ((size_t)NEXP * T_TOK * 4)
#define SCALE_OFF  (PERM_OFF + PERM_BYTES)
#define CNT_OFF    (SCALE_OFF + (size_t)T_TOK * 4)
#define BLKP_OFF   (CNT_OFF + 64)

// ------------------------------------------------------------------
// Kernel 1: gating — logits, softmax, argmax, per-expert compaction.
// One wave per token; also emits xb = bf16(x) while x streams through.
// ------------------------------------------------------------------
__global__ void gate_kernel(const float* __restrict__ x, const float* __restrict__ gw,
                            __bf16* __restrict__ xb, float* __restrict__ scale,
                            int* __restrict__ counts, int* __restrict__ perm,
                            float* __restrict__ blockP) {
  __shared__ float sP[8][NEXP];     // per-wave prob sums (deterministic reduce)
  int tid  = threadIdx.x;
  int wave = tid >> 5, lane = tid & 31;
  int t = blockIdx.x * 8 + wave;

  float acc[NEXP];
#pragma unroll
  for (int e = 0; e < NEXP; ++e) acc[e] = 0.f;

  const float* xr = x + (size_t)t * DIM;
  __bf16* xbr = xb + (size_t)t * DIM;
  for (int k = lane; k < DIM; k += 32) {
    float xv = xr[k];
    xbr[k] = (__bf16)xv;                         // bf16 side-product for GEMM A
    const float* g = gw + (size_t)k * NEXP;
#pragma unroll
    for (int e = 0; e < NEXP; ++e) acc[e] = fmaf(xv, g[e], acc[e]);
  }
#pragma unroll
  for (int e = 0; e < NEXP; ++e) {
#pragma unroll
    for (int m = 16; m >= 1; m >>= 1) acc[e] += __shfl_xor(acc[e], m, 32);
  }

  if (lane == 0) {
    float mx = acc[0]; int am = 0;
#pragma unroll
    for (int e = 1; e < NEXP; ++e) if (acc[e] > mx) { mx = acc[e]; am = e; }  // first-max
    float p[NEXP], s = 0.f;
#pragma unroll
    for (int e = 0; e < NEXP; ++e) { p[e] = __expf(acc[e] - mx); s += p[e]; }
    float inv = 1.f / s;
#pragma unroll
    for (int e = 0; e < NEXP; ++e) { p[e] *= inv; sP[wave][e] = p[e]; }
    scale[t] = p[am];
    int pos = atomicAdd(&counts[am], 1);
    perm[am * T_TOK + pos] = t;
  }
  __syncthreads();
  if (tid < NEXP) {
    float s = 0.f;
#pragma unroll
    for (int w = 0; w < 8; ++w) s += sP[w][tid];
    blockP[blockIdx.x * NEXP + tid] = s;
  }
}

// ------------------------------------------------------------------
// Kernel 2: finalize balance loss + emit num_tokens (int bits).
// ------------------------------------------------------------------
__global__ void loss_kernel(const float* __restrict__ blockP, const int* __restrict__ counts,
                            float* __restrict__ out_loss, int* __restrict__ out_cnt, int nblk) {
  __shared__ float part[NEXP];
  int e = threadIdx.x;
  if (e < NEXP) {
    float s = 0.f;
    for (int b = 0; b < nblk; ++b) s += blockP[b * NEXP + e];
    float P = s / (float)T_TOK;
    float f = (float)counts[e] / (float)T_TOK;
    part[e] = P * f;
    out_cnt[e] = counts[e];
  }
  __syncthreads();
  if (e == 0) {
    float l = 0.f;
#pragma unroll
    for (int i = 0; i < NEXP; ++i) l += part[i];
    *out_loss = (float)NEXP * l;
  }
}

// ------------------------------------------------------------------
// Kernel 3: materialize transposed kron weights in bf16:
//   wT[e][o=b*512+j][d=a*512+i] = sum_p phm[e,p,a,b] * W[e,p,i,j]
// LDS tile transpose: coalesced reads along j, coalesced writes along d.
// ------------------------------------------------------------------
__global__ void kron_kernel(const float* __restrict__ phm, const float* __restrict__ W,
                            __bf16* __restrict__ wT) {
  __shared__ float lds[64 * 65];
  int bid = blockIdx.x;
  int tj = bid & 7;  bid >>= 3;
  int ti = bid & 7;  bid >>= 3;
  int b  = bid & 3;  bid >>= 2;
  int a  = bid & 3;  bid >>= 2;
  int e  = bid;
  int i0 = ti * 64, j0 = tj * 64;

  float s[4];
#pragma unroll
  for (int p = 0; p < 4; ++p) s[p] = phm[((e * 4 + p) * 4 + a) * 4 + b];

  int tid = threadIdx.x;
#pragma unroll
  for (int r = 0; r < 16; ++r) {
    int idx = tid + r * 256;
    int jl = idx & 63, il = idx >> 6;
    float v = 0.f;
#pragma unroll
    for (int p = 0; p < 4; ++p)
      v = fmaf(s[p], W[(((size_t)(e * 4 + p)) * 512 + (i0 + il)) * 512 + (j0 + jl)], v);
    lds[jl * 65 + il] = v;                      // transposed store into LDS
  }
  __syncthreads();
  __bf16* dst = wT + (size_t)e * DIM * ODIM;
#pragma unroll
  for (int r = 0; r < 16; ++r) {
    int idx = tid + r * 256;
    int dl = idx & 63, ol = idx >> 6;
    dst[(size_t)(b * 512 + j0 + ol) * DIM + (a * 512 + i0 + dl)] = (__bf16)lds[ol * 65 + dl];
  }
}

// ------------------------------------------------------------------
// Kernel 4: per-expert gathered GEMM, bf16 WMMA 16x16x32.
// Block = 256 threads = 8 waves (2Mx4N), tile 128(M) x 256(N).
// Wave tile 64x64 = 4x4 WMMA accumulators; 16 WMMA per 16 b128 loads.
// ------------------------------------------------------------------
__global__ void __launch_bounds__(256, 1)
moe_gemm_kernel(const __bf16* __restrict__ xb, const __bf16* __restrict__ wT,
                const int* __restrict__ perm, const int* __restrict__ counts,
                const float* __restrict__ scale, const float* __restrict__ bias,
                float* __restrict__ out) {
  int e   = blockIdx.z;
  int cnt = counts[e];
  int m0  = blockIdx.x * 128;
  if (m0 >= cnt) return;                       // empty tile for this expert
  int n0  = blockIdx.y * 256;

  int tid  = threadIdx.x;
  int wave = tid >> 5, lane = tid & 31;
  int waveM = wave >> 2, waveN = wave & 3;     // 2 x 4 wave grid
  int half = lane >> 4, l15 = lane & 15;
  int rowBase = m0 + waveM * 64;
  int colBase = n0 + waveN * 64;

  const int* permE = perm + e * T_TOK;
  int tok[4];
#pragma unroll
  for (int ms = 0; ms < 4; ++ms) {             // A rows: M = lane&15 per subtile
    int r = rowBase + ms * 16 + l15;
    tok[ms] = permE[r < cnt ? r : cnt - 1];    // clamp; invalid rows never stored
  }

  v8f acc[4][4] = {};
  const __bf16* wTe = wT + (size_t)e * DIM * ODIM;

  union AU { v16bf v; uint4 q[2]; };

#pragma unroll 1
  for (int k0 = 0; k0 < DIM; k0 += 32) {
    // A fragments (bf16): K = j + 16*half -> two 16B chunks
    v16bf A[4];
#pragma unroll
    for (int ms = 0; ms < 4; ++ms) {
      const __bf16* ap = xb + (size_t)tok[ms] * DIM + k0 + half * 16;
      AU u;
      u.q[0] = *(const uint4*)(ap);            // j=0..7  : K = half*16 + 0..7
      u.q[1] = *(const uint4*)(ap + 8);        // j=8..15 : K = half*16 + 8..15
      A[ms] = u.v;
    }
    // B fragments: K = j + 16*half, col = lane&15 (contiguous along d in wT)
    v16bf B[4];
#pragma unroll
    for (int ns = 0; ns < 4; ++ns) {
      const __bf16* bp = wTe + (size_t)(colBase + ns * 16 + l15) * DIM + k0 + 16 * half;
      B[ns] = *(const v16bf*)bp;               // 32-byte aligned contiguous load
    }
#pragma unroll
    for (int ms = 0; ms < 4; ++ms)
#pragma unroll
      for (int ns = 0; ns < 4; ++ns)
        acc[ms][ns] = __builtin_amdgcn_wmma_f32_16x16x32_bf16(
            false, A[ms], false, B[ns], (short)0, acc[ms][ns], false, false);
  }

  // Epilogue: C layout m = r + 8*half, n = lane&15. out = (acc + b) * prob.
  float bcol[4];
#pragma unroll
  for (int ns = 0; ns < 4; ++ns) bcol[ns] = bias[e * ODIM + colBase + ns * 16 + l15];
#pragma unroll
  for (int ms = 0; ms < 4; ++ms) {
#pragma unroll
    for (int r = 0; r < 8; ++r) {
      int rowg = rowBase + ms * 16 + half * 8 + r;
      if (rowg < cnt) {
        int t = permE[rowg];
        float sc = scale[t];
#pragma unroll
        for (int ns = 0; ns < 4; ++ns) {
          int col = colBase + ns * 16 + l15;
          out[(size_t)t * ODIM + col] = (acc[ms][ns][r] + bcol[ns]) * sc;
        }
      }
    }
  }
}

// ------------------------------------------------------------------
extern "C" void kernel_launch(void* const* d_in, const int* in_sizes, int n_in,
                              void* d_out, int out_size, void* d_ws, size_t ws_size,
                              hipStream_t stream) {
  const float* x    = (const float*)d_in[0];   // [4,2048,2048]
  const float* gw   = (const float*)d_in[1];   // [2048,8]
  const float* phm  = (const float*)d_in[2];   // [8,4,4,4]
  const float* W    = (const float*)d_in[3];   // [8,4,512,512]
  const float* bias = (const float*)d_in[4];   // [8,2048]

  char* ws = (char*)d_ws;
  __bf16* wT    = (__bf16*)(ws + WT_OFF);
  __bf16* xb    = (__bf16*)(ws + XB_OFF);
  int*    perm  = (int*)  (ws + PERM_OFF);
  float*  scale = (float*)(ws + SCALE_OFF);
  int*    counts= (int*)  (ws + CNT_OFF);
  float*  blockP= (float*)(ws + BLKP_OFF);

  float* out      = (float*)d_out;
  float* out_loss = out + (size_t)T_TOK * ODIM;              // scalar after [T,O]
  int*   out_cnt  = (int*)(out + (size_t)T_TOK * ODIM + 1);  // int32 bits in tail

  hipMemsetAsync(counts, 0, NEXP * sizeof(int), stream);
  gate_kernel<<<T_TOK / 8, 256, 0, stream>>>(x, gw, xb, scale, counts, perm, blockP);
  loss_kernel<<<1, 32, 0, stream>>>(blockP, counts, out_loss, out_cnt, T_TOK / 8);
  kron_kernel<<<NEXP * 4 * 4 * 8 * 8, 256, 0, stream>>>(phm, W, wT);
  moe_gemm_kernel<<<dim3(T_TOK / 128, ODIM / 256, NEXP), 256, 0, stream>>>(
      xb, wT, perm, counts, scale, bias, out);
}
